// Qwen2EncoderLayer_61787399520463
// MI455X (gfx1250) — matrix operand order repaired and checked by reading, not compiled, not run on hardware
//
#include <hip/hip_runtime.h>
#include <hip/hip_bf16.h>
#include <math.h>

typedef __bf16 bf16;
typedef __attribute__((ext_vector_type(16))) __bf16 v16bf;
typedef __attribute__((ext_vector_type(8)))  __bf16 bf16x8;
typedef __attribute__((ext_vector_type(2)))  __bf16 bf16x2;
typedef __attribute__((ext_vector_type(8)))  float  v8f;

#define HD    2048
#define NB    4
#define SEQ   1024
#define BSR   (NB*SEQ)       // 4096 rows of activations
#define NHEAD 16
#define NKV   4
#define HDIM  128
#define KVDIM (NKV*HDIM)     // 512

// ---------------------------------------------------------------------------
// CDNA5 async global->LDS copy (GLOBAL_LOAD_ASYNC_TO_LDS_B128, ASYNCcnt).
// Generic LDS pointers carry the LDS byte offset in their low 32 bits.
// ---------------------------------------------------------------------------
__device__ __forceinline__ void async_copy_b128(void* lds_dst, const void* gsrc) {
    uint32_t ldsoff = (uint32_t)(uintptr_t)lds_dst;
    asm volatile("global_load_async_to_lds_b128 %0, %1, off"
                 :: "v"(ldsoff), "v"((uint64_t)(uintptr_t)gsrc)
                 : "memory");
}
__device__ __forceinline__ void wait_async0() {
    asm volatile("s_wait_asynccnt 0" ::: "memory");
}

// ---------------------------------------------------------------------------
// WMMA fragment loaders (16x16x32 bf16, contiguous k = 16*(lane/16) + e in A and B)
// ---------------------------------------------------------------------------
__device__ __forceinline__ v16bf load_a_frag(const bf16* base, int stride, int lane) {
    int row = lane & 15;
    int kh  = (lane >> 4) << 4;
    v16bf r;
#pragma unroll
    for (int e = 0; e < 16; ++e) {
        int k = kh + e;
        r[e] = base[(size_t)row * stride + k];
    }
    return r;
}

// element (k,n) at base[k*stride + n]
__device__ __forceinline__ v16bf load_b_frag(const bf16* base, int stride, int lane) {
    int c  = lane & 15;
    int kb = (lane >> 4) << 4;
    v16bf r;
#pragma unroll
    for (int e = 0; e < 16; ++e) r[e] = base[(size_t)(kb + e) * stride + c];
    return r;
}

// element (k,n) at base[n*stride + k] -- per-lane contiguous K (vectorizes)
__device__ __forceinline__ v16bf load_bT_frag(const bf16* base, int stride, int lane) {
    int c  = lane & 15;
    int kb = (lane >> 4) << 4;
    v16bf r;
#pragma unroll
    for (int e = 0; e < 16; ++e) r[e] = base[(size_t)c * stride + kb + e];
    return r;
}

__device__ __forceinline__ v8f wmma_bf16(v16bf a, v16bf b, v8f c) {
    return __builtin_amdgcn_wmma_f32_16x16x32_bf16(false, a, false, b, (short)0, c,
                                                   false, false);
}

// ---------------------------------------------------------------------------
// Elementwise helpers
// ---------------------------------------------------------------------------
__global__ __launch_bounds__(256)
void f32_to_bf16_kernel(const float* __restrict__ in, bf16* __restrict__ out, int n) {
    int i = blockIdx.x * 256 + threadIdx.x;
    if (i < n) out[i] = (bf16)in[i];
}

__global__ __launch_bounds__(256)
void rmsnorm_kernel(const float* __restrict__ x, const float* __restrict__ g,
                    bf16* __restrict__ y, int ncols) {
    int row = blockIdx.x;
    const float* xr = x + (size_t)row * ncols;
    __shared__ float red[256];
    float s = 0.f;
    for (int i = threadIdx.x; i < ncols; i += 256) { float v = xr[i]; s += v * v; }
    red[threadIdx.x] = s;
    __syncthreads();
    for (int off = 128; off > 0; off >>= 1) {
        if (threadIdx.x < off) red[threadIdx.x] += red[threadIdx.x + off];
        __syncthreads();
    }
    float inv = rsqrtf(red[0] / (float)ncols + 1e-8f);
    bf16* yr = y + (size_t)row * ncols;
    for (int i = threadIdx.x; i < ncols; i += 256) yr[i] = (bf16)(xr[i] * inv * g[i]);
}

__global__ __launch_bounds__(256)
void silu_mul_kernel(const float* __restrict__ gate, const float* __restrict__ up,
                     bf16* __restrict__ out, int n) {
    int i = blockIdx.x * 256 + threadIdx.x;
    if (i < n) {
        float gv = gate[i];
        float sg = gv / (1.f + __expf(-gv));
        out[i] = (bf16)(up[i] * sg);
    }
}

// RoPE (interleaved pairs) + [B,S,H,D] -> [B,H,S,D] transpose, f32 -> bf16.
__global__ __launch_bounds__(256)
void rope_transpose_kernel(const float* __restrict__ in, bf16* __restrict__ out,
                           int H, int apply_rope) {
    int p = blockIdx.x * 256 + threadIdx.x;
    int total = NB * SEQ * H * (HDIM / 2);
    if (p >= total) return;
    int d2 = p & 63;
    int t  = p >> 6;
    int h  = t % H;
    int t2 = t / H;
    int s  = t2 % SEQ;
    int b  = t2 / SEQ;
    size_t iofs = (((size_t)(b * SEQ + s) * H + h) * HDIM) + 2 * d2;
    float x0 = in[iofs], x1 = in[iofs + 1];
    float y0 = x0, y1 = x1;
    if (apply_rope) {
        float inv = __powf(10000.0f, -(float)(2 * d2) * (1.0f / 128.0f));
        float ang = (float)s * inv;
        float c = __cosf(ang), sn = __sinf(ang);
        y0 = c * x0 - sn * x1;
        y1 = c * x1 + sn * x0;
    }
    size_t oofs = (((size_t)(b * H + h) * SEQ + s) * HDIM) + 2 * d2;
    out[oofs]     = (bf16)y0;
    out[oofs + 1] = (bf16)y1;
}

// ---------------------------------------------------------------------------
// Tiled WMMA GEMM: out[M,N] = A[M,K](bf16) @ W[K,N](bf16) + bias (+residual)
// BM=64, BN=128, BK=64; 256 threads = 8 waves; wave tile 32x32.
// 8 WMMAs per barrier pair. A tile: async b128 global->LDS. B tile: vector
// b128 global loads stored transposed (Bs[n*72 + k]) so B fragments are
// contiguous per lane (ds_load_b128).
// ---------------------------------------------------------------------------
template <bool HAS_RES>
__global__ __launch_bounds__(256)
void gemm_bf16_kernel(const bf16* __restrict__ A, const bf16* __restrict__ W,
                      const float* __restrict__ bias, const float* __restrict__ residual,
                      float* __restrict__ out, int M, int N, int K) {
    __shared__ bf16 As[64 * 72];     // 64 rows x 64 k, padded stride 72
    __shared__ bf16 Bs[128 * 72];    // transposed: [n 0..127][k 0..63], padded
    const int m0   = blockIdx.y * 64;
    const int n0   = blockIdx.x * 128;
    const int tid  = threadIdx.x;
    const int lane = tid & 31;
    const int wid  = tid >> 5;
    const int wm   = (wid & 1) * 32;
    const int wn   = (wid >> 1) * 32;

    const int ar = tid >> 2;          // 0..63
    const int ac = (tid & 3) * 8;     // 0,8,16,24  (+32 for second half)
    const int br = (tid >> 4) * 2;    // k row pair base 0..30 (+32 second half)
    const int bc = (tid & 15) * 8;    // col chunk 0..120

    v8f acc[2][2];
#pragma unroll
    for (int i = 0; i < 2; ++i)
#pragma unroll
        for (int j = 0; j < 2; ++j)
#pragma unroll
            for (int r = 0; r < 8; ++r) acc[i][j][r] = 0.f;

    for (int k0 = 0; k0 < K; k0 += 64) {
        // A tile: two async b128 copies per thread
        async_copy_b128(&As[ar * 72 + ac],      &A[(size_t)(m0 + ar) * K + k0 + ac]);
        async_copy_b128(&As[ar * 72 + ac + 32], &A[(size_t)(m0 + ar) * K + k0 + ac + 32]);
        // B tile: 4 vector loads, transposed packed b32 stores
#pragma unroll
        for (int hh = 0; hh < 2; ++hh) {
            int kr = br + hh * 32;
            bf16x8 w0 = *(const bf16x8*)&W[(size_t)(k0 + kr) * N + n0 + bc];
            bf16x8 w1 = *(const bf16x8*)&W[(size_t)(k0 + kr + 1) * N + n0 + bc];
#pragma unroll
            for (int j = 0; j < 8; ++j) {
                bf16x2 p; p[0] = w0[j]; p[1] = w1[j];
                *(bf16x2*)&Bs[(bc + j) * 72 + kr] = p;
            }
        }
        wait_async0();
        __syncthreads();

#pragma unroll
        for (int kk = 0; kk < 2; ++kk) {
            v16bf af[2], bfm[2];
#pragma unroll
            for (int mi = 0; mi < 2; ++mi)
                af[mi] = load_a_frag(&As[(wm + mi * 16) * 72 + kk * 32], 72, lane);
#pragma unroll
            for (int ni = 0; ni < 2; ++ni)
                bfm[ni] = load_bT_frag(&Bs[(wn + ni * 16) * 72 + kk * 32], 72, lane);
#pragma unroll
            for (int mi = 0; mi < 2; ++mi)
#pragma unroll
                for (int ni = 0; ni < 2; ++ni)
                    acc[mi][ni] = wmma_bf16(af[mi], bfm[ni], acc[mi][ni]);
        }
        __syncthreads();
    }

#pragma unroll
    for (int mi = 0; mi < 2; ++mi)
#pragma unroll
        for (int ni = 0; ni < 2; ++ni)
#pragma unroll
            for (int r = 0; r < 8; ++r) {
                int grow = m0 + wm + mi * 16 + r + ((lane >> 4) << 3);
                int gcol = n0 + wn + ni * 16 + (lane & 15);
                size_t o = (size_t)grow * N + gcol;
                float v = acc[mi][ni][r] + bias[gcol];
                if (HAS_RES) v += residual[o];
                out[o] = v;
            }
}

// ---------------------------------------------------------------------------
// Flash attention, one wave per (b, h, 16-row q tile). GQA: kv head = h/4.
// V tile (32x128) staged via async b128 copies, overlapped with score WMMAs.
// ---------------------------------------------------------------------------
__global__ __launch_bounds__(32)
void attn_kernel(const bf16* __restrict__ qm, const bf16* __restrict__ km,
                 const bf16* __restrict__ vm, const float* __restrict__ seq_mask,
                 bf16* __restrict__ ctx) {
    __shared__ bf16 Ps[16 * 40];
    __shared__ bf16 Vs[32 * 136];
    const int lane = threadIdx.x & 31;
    const int qt   = blockIdx.x & 63;
    const int h    = (blockIdx.x >> 6) & 15;
    const int b    = blockIdx.x >> 10;
    const int kvh  = h >> 2;
    const int q0   = qt * 16;
    const int half = lane >> 4;
    const int lc   = lane & 15;
    const float scale = 0.08838834764831845f;    // 1/sqrt(128)

    const bf16* qbase = qm + ((size_t)(b * NHEAD + h) * SEQ + q0) * HDIM;
    const bf16* kbase = km + ((size_t)(b * NKV + kvh) * SEQ) * HDIM;
    const bf16* vbase = vm + ((size_t)(b * NKV + kvh) * SEQ) * HDIM;

    v16bf qf[4];
#pragma unroll
    for (int kk = 0; kk < 4; ++kk) qf[kk] = load_a_frag(qbase + kk * 32, HDIM, lane);

    float mrow[8], lrow[8], smq[8];
    v8f cacc[8];
#pragma unroll
    for (int r = 0; r < 8; ++r) {
        mrow[r] = -1e30f;
        lrow[r] = 0.f;
        smq[r]  = seq_mask[(size_t)b * SEQ + q0 + r + half * 8];
    }
#pragma unroll
    for (int d = 0; d < 8; ++d)
#pragma unroll
        for (int r = 0; r < 8; ++r) cacc[d][r] = 0.f;

    for (int kt = 0; kt < SEQ; kt += 32) {
        const bf16* vrow = vbase + (size_t)(kt + lane) * HDIM;
#pragma unroll
        for (int j = 0; j < 16; ++j)
            async_copy_b128(&Vs[lane * 136 + j * 8], vrow + j * 8);

        v8f sacc[2];
#pragma unroll
        for (int nt = 0; nt < 2; ++nt)
#pragma unroll
            for (int r = 0; r < 8; ++r) sacc[nt][r] = 0.f;
#pragma unroll
        for (int nt = 0; nt < 2; ++nt)
#pragma unroll
            for (int kk = 0; kk < 4; ++kk) {
                v16bf kf = load_bT_frag(kbase + (size_t)(kt + nt * 16) * HDIM + kk * 32,
                                        HDIM, lane);
                sacc[nt] = wmma_bf16(qf[kk], kf, sacc[nt]);
            }

        float smk0 = seq_mask[(size_t)b * SEQ + kt + lc];
        float smk1 = seq_mask[(size_t)b * SEQ + kt + 16 + lc];

#pragma unroll
        for (int r = 0; r < 8; ++r) {
            float s0 = sacc[0][r] * scale;
            float s1 = sacc[1][r] * scale;
            if (smq[r] * smk0 == 0.f) s0 = -1e30f;
            if (smq[r] * smk1 == 0.f) s1 = -1e30f;
            float tm = fmaxf(s0, s1);
            tm = fmaxf(tm, __shfl_xor(tm, 1));
            tm = fmaxf(tm, __shfl_xor(tm, 2));
            tm = fmaxf(tm, __shfl_xor(tm, 4));
            tm = fmaxf(tm, __shfl_xor(tm, 8));
            float mnew = fmaxf(mrow[r], tm);
            float p0 = __expf(s0 - mnew);
            float p1 = __expf(s1 - mnew);
            float rs = p0 + p1;
            rs += __shfl_xor(rs, 1);
            rs += __shfl_xor(rs, 2);
            rs += __shfl_xor(rs, 4);
            rs += __shfl_xor(rs, 8);
            float corr = __expf(mrow[r] - mnew);
            lrow[r] = lrow[r] * corr + rs;
            mrow[r] = mnew;
#pragma unroll
            for (int d = 0; d < 8; ++d) cacc[d][r] *= corr;
            Ps[(r + half * 8) * 40 + lc]      = (bf16)p0;
            Ps[(r + half * 8) * 40 + 16 + lc] = (bf16)p1;
        }
        __syncthreads();
        wait_async0();
        v16bf pf = load_a_frag(Ps, 40, lane);
#pragma unroll
        for (int d = 0; d < 8; ++d) {
            v16bf vf = load_b_frag(&Vs[d * 16], 136, lane);
            cacc[d] = wmma_bf16(pf, vf, cacc[d]);
        }
        __syncthreads();
    }

#pragma unroll
    for (int d = 0; d < 8; ++d)
#pragma unroll
        for (int r = 0; r < 8; ++r) {
            float l = lrow[r];
            float v = (l > 0.f) ? cacc[d][r] / l : 0.f;
            int row = q0 + r + half * 8;
            int col = h * HDIM + d * 16 + lc;
            ctx[(size_t)(b * SEQ + row) * HD + col] = (bf16)v;
        }
}

// ---------------------------------------------------------------------------
// Host-side orchestration
// ---------------------------------------------------------------------------
extern "C" void kernel_launch(void* const* d_in, const int* in_sizes, int n_in,
                              void* d_out, int out_size, void* d_ws, size_t ws_size,
                              hipStream_t stream) {
    (void)in_sizes; (void)n_in; (void)out_size; (void)ws_size;
    const float* enc   = (const float*)d_in[0];
    const float* smask = (const float*)d_in[1];
    const float* g_in  = (const float*)d_in[4];
    const float* g_ffn = (const float*)d_in[5];
    const float* w_q = (const float*)d_in[6];  const float* b_q = (const float*)d_in[7];
    const float* w_k = (const float*)d_in[8];  const float* b_k = (const float*)d_in[9];
    const float* w_v = (const float*)d_in[10]; const float* b_v = (const float*)d_in[11];
    const float* w_o = (const float*)d_in[12]; const float* b_o = (const float*)d_in[13];
    const float* w_g = (const float*)d_in[14]; const float* b_g = (const float*)d_in[15];
    const float* w_u = (const float*)d_in[16]; const float* b_u = (const float*)d_in[17];
    const float* w_d = (const float*)d_in[18]; const float* b_d = (const float*)d_in[19];
    float* outp = (float*)d_out;

    char* ws = (char*)d_ws;
    size_t off = 0;
    auto alloc = [&](size_t bytes) {
        char* p = ws + off;
        off += (bytes + 255) & ~(size_t)255;
        return p;
    };
    bf16* wq_b = (bf16*)alloc((size_t)HD * HD * 2);
    bf16* wk_b = (bf16*)alloc((size_t)HD * KVDIM * 2);
    bf16* wv_b = (bf16*)alloc((size_t)HD * KVDIM * 2);
    bf16* wo_b = (bf16*)alloc((size_t)HD * HD * 2);
    bf16* wg_b = (bf16*)alloc((size_t)HD * HD * 2);
    bf16* wu_b = (bf16*)alloc((size_t)HD * HD * 2);
    bf16* wd_b = (bf16*)alloc((size_t)HD * HD * 2);
    bf16*  xnorm = (bf16*)alloc((size_t)BSR * HD * 2);          // reused as h
    float* qf    = (float*)alloc((size_t)BSR * HD * 4);         // reused as gate
    char*  regD  = alloc((size_t)BSR * KVDIM * 4 * 2);          // k+v f32, reused as t
    float* kf    = (float*)regD;
    float* vf    = (float*)(regD + (size_t)BSR * KVDIM * 4);
    char*  regE  = alloc((size_t)BSR * HD * 4);                 // q/k/v bhsd, reused as up
    bf16*  q_t   = (bf16*)regE;
    bf16*  k_t   = (bf16*)(regE + (size_t)BSR * HD * 2);
    bf16*  v_t   = (bf16*)(regE + (size_t)BSR * HD * 2 + (size_t)BSR * KVDIM * 2);
    bf16*  ctxb  = (bf16*)alloc((size_t)BSR * HD * 2);
    float* attn_out = (float*)alloc((size_t)BSR * HD * 4);

    bf16*  hbuf  = xnorm;
    float* gatef = qf;
    float* upf   = (float*)regE;
    bf16*  tbuf  = (bf16*)regD;

    const int nHD2 = HD * HD;
    const int nHK  = HD * KVDIM;
    f32_to_bf16_kernel<<<(nHD2 + 255) / 256, 256, 0, stream>>>(w_q, wq_b, nHD2);
    f32_to_bf16_kernel<<<(nHK  + 255) / 256, 256, 0, stream>>>(w_k, wk_b, nHK);
    f32_to_bf16_kernel<<<(nHK  + 255) / 256, 256, 0, stream>>>(w_v, wv_b, nHK);
    f32_to_bf16_kernel<<<(nHD2 + 255) / 256, 256, 0, stream>>>(w_o, wo_b, nHD2);
    f32_to_bf16_kernel<<<(nHD2 + 255) / 256, 256, 0, stream>>>(w_g, wg_b, nHD2);
    f32_to_bf16_kernel<<<(nHD2 + 255) / 256, 256, 0, stream>>>(w_u, wu_b, nHD2);
    f32_to_bf16_kernel<<<(nHD2 + 255) / 256, 256, 0, stream>>>(w_d, wd_b, nHD2);

    rmsnorm_kernel<<<BSR, 256, 0, stream>>>(enc, g_in, xnorm, HD);

    dim3 gFull(HD / 128, BSR / 64);
    dim3 gKV(KVDIM / 128, BSR / 64);
    gemm_bf16_kernel<false><<<gFull, 256, 0, stream>>>(xnorm, wq_b, b_q, nullptr, qf, BSR, HD, HD);
    gemm_bf16_kernel<false><<<gKV,   256, 0, stream>>>(xnorm, wk_b, b_k, nullptr, kf, BSR, KVDIM, HD);
    gemm_bf16_kernel<false><<<gKV,   256, 0, stream>>>(xnorm, wv_b, b_v, nullptr, vf, BSR, KVDIM, HD);

    int nq = NB * SEQ * NHEAD * (HDIM / 2);
    int nk = NB * SEQ * NKV * (HDIM / 2);
    rope_transpose_kernel<<<(nq + 255) / 256, 256, 0, stream>>>(qf, q_t, NHEAD, 1);
    rope_transpose_kernel<<<(nk + 255) / 256, 256, 0, stream>>>(kf, k_t, NKV, 1);
    rope_transpose_kernel<<<(nk + 255) / 256, 256, 0, stream>>>(vf, v_t, NKV, 0);

    attn_kernel<<<NB * NHEAD * (SEQ / 16), 32, 0, stream>>>(q_t, k_t, v_t, smask, ctxb);

    gemm_bf16_kernel<true><<<gFull, 256, 0, stream>>>(ctxb, wo_b, b_o, enc, attn_out, BSR, HD, HD);

    rmsnorm_kernel<<<BSR, 256, 0, stream>>>(attn_out, g_ffn, hbuf, HD);

    gemm_bf16_kernel<false><<<gFull, 256, 0, stream>>>(hbuf, wg_b, b_g, nullptr, gatef, BSR, HD, HD);
    gemm_bf16_kernel<false><<<gFull, 256, 0, stream>>>(hbuf, wu_b, b_u, nullptr, upf, BSR, HD, HD);
    int nact = BSR * HD;
    silu_mul_kernel<<<(nact + 255) / 256, 256, 0, stream>>>(gatef, upf, tbuf, nact);

    gemm_bf16_kernel<true><<<gFull, 256, 0, stream>>>(tbuf, wd_b, b_d, attn_out, outp, BSR, HD, HD);
}
